// EvidentialGNN_19859928777443
// MI455X (gfx1250) — matrix operand order repaired and checked by reading, not compiled, not run on hardware
//
#include <hip/hip_runtime.h>
#include <hip/hip_bf16.h>
#include <math.h>

// ---------- types ----------
typedef __attribute__((ext_vector_type(16))) __bf16         v16bf;
typedef __attribute__((ext_vector_type(8)))  float          v8f;
typedef __attribute__((ext_vector_type(8)))  unsigned short v8us;

union BF16x16 { v16bf v; unsigned short u[16]; };

// f32 pair -> packed bf16 dword; plain casts let clang pick the native
// v_cvt_*bf16_f32 path on gfx1250 (falls back to RNE integer lowering otherwise)
__device__ __forceinline__ unsigned int pack2(float lo, float hi) {
    __bf16 a = (__bf16)lo;
    __bf16 b = (__bf16)hi;
    unsigned int ua = (unsigned int)__builtin_bit_cast(unsigned short, a);
    unsigned int ub = (unsigned int)__builtin_bit_cast(unsigned short, b);
    return ua | (ub << 16);
}

// guaranteed-native f32 scatter atomic (no CAS loop)
__device__ __forceinline__ void atomic_add_f32(float* p, float v) {
    asm volatile("global_atomic_add_f32 %0, %1, off" : : "v"(p), "v"(v) : "memory");
}

// ---------- degree / normalization ----------
__global__ void k_deg_init(float* deg, int n) {
    int i = blockIdx.x * blockDim.x + threadIdx.x;
    if (i < n) deg[i] = 1.0f;           // self-loop
}
__global__ void k_deg_count(const long long* __restrict__ dst, float* deg, int e) {
    int i = blockIdx.x * blockDim.x + threadIdx.x;
    if (i < e) atomic_add_f32(&deg[(int)dst[i]], 1.0f);
}
__global__ void k_deg_rsqrt(float* deg, int n) {
    int i = blockIdx.x * blockDim.x + threadIdx.x;
    if (i < n) deg[i] = rsqrtf(deg[i]); // deg >= 1 always
}

// ---------- bf16 WMMA GEMM: C[M,Ncols] = A[M,Ktot] * W[Ktot,Ncols] ----------
// block = 256 threads = 8 waves; block tile 64x64; wave tile 32x16 (2 WMMA accs)
// double-buffered LDS, one barrier per K-step of 32.
#define BM 64
#define BN 64
#define BK 32
#define LDT 40   // LDS row stride in bf16 elems (80B: 16B-aligned, bank-spread)

__global__ void __launch_bounds__(256)
k_gemm_bf16(const float* __restrict__ A, const float* __restrict__ W,
            float* __restrict__ C, int M, int Ncols, int Ktot) {
    __shared__ __align__(16) unsigned short sA[2][BM * LDT];
    __shared__ __align__(16) unsigned short sB[2][BN * LDT]; // transposed: [n][k]

    const int tid   = threadIdx.x;
    const int wave  = tid >> 5;
    const int lane  = tid & 31;
    const int waveM = wave >> 2;      // 0..1 -> rows waveM*32 .. +31
    const int waveN = wave & 3;       // 0..3 -> cols waveN*16 .. +15
    const int half  = lane >> 4;
    const int r     = lane & 15;

    const int rowBase = blockIdx.x * BM;
    const int colBase = blockIdx.y * BN;

    // A staging: thread covers rows {tid>>3, (tid>>3)+32}, 4 consecutive K
    const int aRow = tid >> 3;            // 0..31
    const int aK   = (tid & 7) << 2;      // 0,4,...,28
    // B staging: fixed col, 8 consecutive K (lane-coalesced global reads)
    const int bN   = tid & 63;            // 0..63
    const int bK   = (tid >> 6) << 3;     // 0,8,16,24

    v8f acc0 = {}, acc1 = {};

    const int nk = Ktot / BK;

    // ---- prologue stage into buffer 0 ----
    {
        #pragma unroll
        for (int it = 0; it < 2; ++it) {
            int row = aRow + it * 32;
            int gr  = rowBase + row;
            float4 v = make_float4(0.f, 0.f, 0.f, 0.f);
            if (gr < M) v = *(const float4*)(A + (size_t)gr * Ktot + aK);
            *(uint2*)&sA[0][row * LDT + aK] = make_uint2(pack2(v.x, v.y), pack2(v.z, v.w));
        }
        const float* wp = W + (size_t)bK * Ncols + colBase + bN;
        float f0 = wp[0];
        float f1 = wp[(size_t)1 * Ncols];
        float f2 = wp[(size_t)2 * Ncols];
        float f3 = wp[(size_t)3 * Ncols];
        float f4 = wp[(size_t)4 * Ncols];
        float f5 = wp[(size_t)5 * Ncols];
        float f6 = wp[(size_t)6 * Ncols];
        float f7 = wp[(size_t)7 * Ncols];
        uint4 pk = make_uint4(pack2(f0, f1), pack2(f2, f3), pack2(f4, f5), pack2(f6, f7));
        *(uint4*)&sB[0][bN * LDT + bK] = pk;
    }
    __syncthreads();

    for (int kb = 0; kb < nk; ++kb) {
        const int cur = kb & 1;
        const unsigned short* pA = sA[cur];
        const unsigned short* pB = sB[cur];

        // ---- issue next-tile global loads early (latency hidden by WMMA) ----
        float4 av0 = make_float4(0.f, 0.f, 0.f, 0.f);
        float4 av1 = make_float4(0.f, 0.f, 0.f, 0.f);
        float  bf0=0,bf1=0,bf2=0,bf3=0,bf4=0,bf5=0,bf6=0,bf7=0;
        const bool have_next = (kb + 1) < nk;
        if (have_next) {
            const int k0 = (kb + 1) * BK;
            int gr0 = rowBase + aRow;
            int gr1 = gr0 + 32;
            if (gr0 < M) av0 = *(const float4*)(A + (size_t)gr0 * Ktot + k0 + aK);
            if (gr1 < M) av1 = *(const float4*)(A + (size_t)gr1 * Ktot + k0 + aK);
            const float* wp = W + (size_t)(k0 + bK) * Ncols + colBase + bN;
            bf0 = wp[0];
            bf1 = wp[(size_t)1 * Ncols];
            bf2 = wp[(size_t)2 * Ncols];
            bf3 = wp[(size_t)3 * Ncols];
            bf4 = wp[(size_t)4 * Ncols];
            bf5 = wp[(size_t)5 * Ncols];
            bf6 = wp[(size_t)6 * Ncols];
            bf7 = wp[(size_t)7 * Ncols];
        }
        if (kb + 2 < nk) {
            __builtin_prefetch(A + (size_t)(rowBase + aRow) * Ktot + (kb + 2) * BK + aK, 0, 1);
        }

        // ---- fragment loads from LDS (ds_load_b128 pairs) ----
        const int ar0 = waveM * 32 + r;
        const int ar1 = ar0 + 16;
        const int bc  = waveN * 16 + r;
        v8us a0lo = *(const v8us*)&pA[ar0 * LDT + 8 * half];
        v8us a0hi = *(const v8us*)&pA[ar0 * LDT + 16 + 8 * half];
        v8us a1lo = *(const v8us*)&pA[ar1 * LDT + 8 * half];
        v8us a1hi = *(const v8us*)&pA[ar1 * LDT + 16 + 8 * half];
        v8us blo  = *(const v8us*)&pB[bc * LDT + 8 * half];
        v8us bhi  = *(const v8us*)&pB[bc * LDT + 16 + 8 * half];

        BF16x16 fa0, fa1, fb;
        #pragma unroll
        for (int i = 0; i < 8; ++i) {
            fa0.u[i] = a0lo[i]; fa0.u[8 + i] = a0hi[i];
            fa1.u[i] = a1lo[i]; fa1.u[8 + i] = a1hi[i];
            fb.u[i]  = blo[i];  fb.u[8 + i]  = bhi[i];
        }

        acc0 = __builtin_amdgcn_wmma_f32_16x16x32_bf16(false, fa0.v, false, fb.v,
                                                       (short)0, acc0, false, false);
        acc1 = __builtin_amdgcn_wmma_f32_16x16x32_bf16(false, fa1.v, false, fb.v,
                                                       (short)0, acc1, false, false);

        // ---- store next tile into the other buffer ----
        if (have_next) {
            const int nxt = cur ^ 1;
            *(uint2*)&sA[nxt][aRow * LDT + aK] =
                make_uint2(pack2(av0.x, av0.y), pack2(av0.z, av0.w));
            *(uint2*)&sA[nxt][(aRow + 32) * LDT + aK] =
                make_uint2(pack2(av1.x, av1.y), pack2(av1.z, av1.w));
            *(uint4*)&sB[nxt][bN * LDT + bK] =
                make_uint4(pack2(bf0, bf1), pack2(bf2, bf3), pack2(bf4, bf5), pack2(bf6, bf7));
        }
        __syncthreads();
    }

    // ---- epilogue: C/D layout: VGPR j -> row 8*half + j, col = lane%16 ----
    const int col  = colBase + waveN * 16 + r;
    const int row0 = rowBase + waveM * 32 + 8 * half;
    #pragma unroll
    for (int j = 0; j < 8; ++j) {
        int row = row0 + j;
        if (row < M) C[(size_t)row * Ncols + col] = acc0[j];
    }
    #pragma unroll
    for (int j = 0; j < 8; ++j) {
        int row = row0 + 16 + j;
        if (row < M) C[(size_t)row * Ncols + col] = acc1[j];
    }
}

// ---------- aggregation ----------
template <int F>
__global__ void k_self_init(const float* __restrict__ h, const float* __restrict__ dinv,
                            float* __restrict__ agg, int n) {
    int t = blockIdx.x * blockDim.x + threadIdx.x;
    if (t < n * F) {
        int node = t / F;
        float w = dinv[node];
        agg[t] = h[t] * w * w;          // self-loop contribution
    }
}

template <int F>
__global__ void k_scatter(const float* __restrict__ h, const float* __restrict__ dinv,
                          const long long* __restrict__ src, const long long* __restrict__ dst,
                          float* __restrict__ agg, int e) {
    long long t = (long long)blockIdx.x * blockDim.x + threadIdx.x;
    int ed = (int)(t / F);
    int f  = (int)(t % F);
    if (ed >= e) return;
    int s = (int)src[ed];
    int d = (int)dst[ed];
    float w = dinv[s] * dinv[d];
    atomic_add_f32(&agg[(size_t)d * F + f], h[(size_t)s * F + f] * w);
}

__global__ void k_bias_relu(const float* __restrict__ agg, const float* __restrict__ b,
                            float* __restrict__ out, int n, int F) {
    int t = blockIdx.x * blockDim.x + threadIdx.x;
    if (t < n * F) {
        float v = agg[t] + b[t % F];
        out[t] = v > 0.f ? v : 0.f;
    }
}

__global__ void k_bias_softplus(const float* __restrict__ agg, const float* __restrict__ b,
                                float* __restrict__ out, int n, int F) {
    int t = blockIdx.x * blockDim.x + threadIdx.x;
    if (t < n * F) {
        float v = agg[t] + b[t % F];
        out[t] = (v > 20.f) ? v : log1pf(__expf(v));
    }
}

// ---------- launcher ----------
extern "C" void kernel_launch(void* const* d_in, const int* in_sizes, int n_in,
                              void* d_out, int out_size, void* d_ws, size_t ws_size,
                              hipStream_t stream) {
    const int IN = 512, HID = 256, CLS = 64;
    const float*     x   = (const float*)d_in[0];
    const long long* ei  = (const long long*)d_in[1];
    const float*     W1  = (const float*)d_in[2];
    const float*     b1  = (const float*)d_in[3];
    const float*     W2  = (const float*)d_in[4];
    const float*     b2  = (const float*)d_in[5];

    const int N = in_sizes[0] / IN;     // 50000
    const int E = in_sizes[1] / 2;      // 800000
    const long long* src = ei;
    const long long* dst = ei + E;

    float* out_ev = (float*)d_out;                  // [N, CLS] evidence
    float* out_h  = out_ev + (size_t)N * CLS;       // [N, HID] hidden (2nd output)

    float* ws      = (float*)d_ws;
    size_t nal     = ((size_t)N + 255) & ~(size_t)255;
    float* dinv    = ws;                            // N
    float* bufH    = ws + nal;                      // N*HID  (h1, later h2)
    float* bufAgg  = bufH + (size_t)N * HID;        // N*HID  (agg1, later agg2)

    const int TB = 256;
    // 1) normalization
    k_deg_init<<<(N + TB - 1) / TB, TB, 0, stream>>>(dinv, N);
    k_deg_count<<<(E + TB - 1) / TB, TB, 0, stream>>>(dst, dinv, E);
    k_deg_rsqrt<<<(N + TB - 1) / TB, TB, 0, stream>>>(dinv, N);

    // 2) GEMM1: h1 = x @ W1   [N,512]x[512,256]
    dim3 g1((N + BM - 1) / BM, HID / BN);
    k_gemm_bf16<<<g1, TB, 0, stream>>>(x, W1, bufH, N, HID, IN);

    // 3) aggregation layer 1
    k_self_init<HID><<<(int)(((size_t)N * HID + TB - 1) / TB), TB, 0, stream>>>(bufH, dinv, bufAgg, N);
    k_scatter<HID><<<(int)(((size_t)E * HID + TB - 1) / TB), TB, 0, stream>>>(bufH, dinv, src, dst, bufAgg, E);
    k_bias_relu<<<(int)(((size_t)N * HID + TB - 1) / TB), TB, 0, stream>>>(bufAgg, b1, out_h, N, HID);

    // 4) GEMM2: h2 = h @ W2   [N,256]x[256,64]  (reuse bufH for h2)
    dim3 g2((N + BM - 1) / BM, CLS / BN);
    k_gemm_bf16<<<g2, TB, 0, stream>>>(out_h, W2, bufH, N, CLS, HID);

    // 5) aggregation layer 2 + softplus (reuse bufAgg)
    k_self_init<CLS><<<(int)(((size_t)N * CLS + TB - 1) / TB), TB, 0, stream>>>(bufH, dinv, bufAgg, N);
    k_scatter<CLS><<<(int)(((size_t)E * CLS + TB - 1) / TB), TB, 0, stream>>>(bufH, dinv, src, dst, bufAgg, E);
    k_bias_softplus<<<(int)(((size_t)N * CLS + TB - 1) / TB), TB, 0, stream>>>(bufAgg, b2, out_ev, N, CLS);
}
